// NET_Wrapper_66357244723409
// MI455X (gfx1250) — compile-verified, matrix operand order checked by reading, not run on hardware
//
#include <hip/hip_runtime.h>
#include <hip/hip_bf16.h>

typedef __attribute__((ext_vector_type(16))) _Float16 v16h;
typedef __attribute__((ext_vector_type(8)))  _Float16 v8h;
typedef __attribute__((ext_vector_type(8)))  float    v8f;

#define FRAME 320
#define SHIFT 160
#define FDIM  256
#define HDIM  256
#define NCLS  101
#define NB    32
#define TIN   1000
#define TA    400
#define LSIG  160000
#define LAUX  64000

// ---------------------------------------------------------------------------
// Framing: signal [NB, L] -> f16 frames [NB*T, FRAME], zero-padded tail.
// ---------------------------------------------------------------------------
__global__ void k_frames(const float* __restrict__ sig, _Float16* __restrict__ dst,
                         int T, int L) {
    long i = (long)blockIdx.x * blockDim.x + threadIdx.x;
    long total = (long)NB * T * FRAME;
    if (i >= total) return;
    int s = (int)(i % FRAME);
    long bt = i / FRAME;
    int t = (int)(bt % T);
    int b = (int)(bt / T);
    long idx = (long)t * SHIFT + s;
    float v = (idx < L) ? sig[(long)b * L + idx] : 0.f;
    dst[i] = (_Float16)v;
}

// ---------------------------------------------------------------------------
// Weight transpose + f32->f16:  W[K,N] row-major -> Wt[N,K] f16
// ---------------------------------------------------------------------------
__global__ void k_transpose_f16(const float* __restrict__ W, _Float16* __restrict__ Wt,
                                int K, int N) {
    int i = blockIdx.x * blockDim.x + threadIdx.x;
    if (i >= K * N) return;
    int k = i % K, n = i / K;
    Wt[(long)n * K + k] = (_Float16)W[(long)k * N + n];
}

// ---------------------------------------------------------------------------
// Fragment load: 16-bit A/B layout for v_wmma_f32_16x16x32_f16.
// element e of lane L maps to k = (e/8)*16 + (L/16)*8 + e%8.
// ---------------------------------------------------------------------------
__device__ __forceinline__ v16h load_frag(const _Float16* __restrict__ p,
                                          int k0, int half) {
    v8h a0 = *(const v8h*)(p + k0 + half * 8);
    v8h a1 = *(const v8h*)(p + k0 + 16 + half * 8);
    v16h f;
#pragma unroll
    for (int e = 0; e < 8; ++e) { f[e] = a0[e]; f[e + 8] = a1[e]; }
    return f;
}

// ---------------------------------------------------------------------------
// Generic WMMA GEMM: C[M,N] = A[M,K](f16) * Bt[N,K](f16)^T  + epilogue
// One wave computes a 32x64 tile: 2 A-fragments x 4 B-fragments -> 8 WMMAs
// per 32-wide k-step. K-loop is unrolled x2 with true double buffering
// (distinct stage-A / stage-B fragment registers refilled by loads, never
// copied), so the steady state is: load(B) | wmma(A) | load(A') | wmma(B).
// Requires K % 64 == 0 (all call sites: K = 320 or 256).
// Epilogue modes:
//   0: v+bias          -> Cf32, Cf16
//   1: relu(v+bias)    -> Cf32, Cf16
//   2: tanh(relu(v+rowbias)*gamma+beta) -> Cf16
//   3: v+bias          -> Cf32
//   4: Cf32 += tanh(v+rowbias) (in place), result also -> Cf16
// ---------------------------------------------------------------------------
__launch_bounds__(256)
__global__ void k_gemm_wmma(const _Float16* __restrict__ A,
                            const _Float16* __restrict__ Bt,
                            int M, int N, int K,
                            const float* __restrict__ bias,
                            const float* __restrict__ rowbias, int rowsPerBatch,
                            const float* __restrict__ gamma,
                            const float* __restrict__ beta,
                            float* __restrict__ Cf32,
                            _Float16* __restrict__ Cf16,
                            int mode) {
    int wave = (blockIdx.x * blockDim.x + threadIdx.x) >> 5;
    int lane = threadIdx.x & 31;
    int ngroups = N >> 6;                 // groups of 64 columns
    int mt = wave / ngroups;              // 32-row tiles
    int ng = wave % ngroups;
    if (mt * 32 >= M) return;

    int half = lane >> 4;                 // 0 or 1
    int l15  = lane & 15;

    const _Float16* pa0 = A + (long)(mt * 32 + l15) * K;
    const _Float16* pa1 = A + (long)(mt * 32 + 16 + l15) * K;
    const _Float16* pb[4];
#pragma unroll
    for (int j = 0; j < 4; ++j)
        pb[j] = Bt + (long)(ng * 64 + j * 16 + l15) * K;

    v8f acc[2][4];
#pragma unroll
    for (int mi = 0; mi < 2; ++mi)
#pragma unroll
        for (int j = 0; j < 4; ++j)
#pragma unroll
            for (int e = 0; e < 8; ++e) acc[mi][j][e] = 0.f;

    // stage-A fragments (k0), stage-B fragments (k0+32)
    v16h a0A = load_frag(pa0, 0, half);
    v16h a1A = load_frag(pa1, 0, half);
    v16h bA0 = load_frag(pb[0], 0, half);
    v16h bA1 = load_frag(pb[1], 0, half);
    v16h bA2 = load_frag(pb[2], 0, half);
    v16h bA3 = load_frag(pb[3], 0, half);
    v16h a0B, a1B, bB0, bB1, bB2, bB3;

#define WMMA_STEP(A0, A1, B0, B1, B2, B3)                                        \
    do {                                                                         \
        acc[0][0] = __builtin_amdgcn_wmma_f32_16x16x32_f16(                      \
            false, A0, false, B0, (short)0, acc[0][0], false, false);            \
        acc[1][0] = __builtin_amdgcn_wmma_f32_16x16x32_f16(                      \
            false, A1, false, B0, (short)0, acc[1][0], false, false);            \
        acc[0][1] = __builtin_amdgcn_wmma_f32_16x16x32_f16(                      \
            false, A0, false, B1, (short)0, acc[0][1], false, false);            \
        acc[1][1] = __builtin_amdgcn_wmma_f32_16x16x32_f16(                      \
            false, A1, false, B1, (short)0, acc[1][1], false, false);            \
        acc[0][2] = __builtin_amdgcn_wmma_f32_16x16x32_f16(                      \
            false, A0, false, B2, (short)0, acc[0][2], false, false);            \
        acc[1][2] = __builtin_amdgcn_wmma_f32_16x16x32_f16(                      \
            false, A1, false, B2, (short)0, acc[1][2], false, false);            \
        acc[0][3] = __builtin_amdgcn_wmma_f32_16x16x32_f16(                      \
            false, A0, false, B3, (short)0, acc[0][3], false, false);            \
        acc[1][3] = __builtin_amdgcn_wmma_f32_16x16x32_f16(                      \
            false, A1, false, B3, (short)0, acc[1][3], false, false);            \
    } while (0)

    int k0 = 0;
    for (; k0 < K - 64; k0 += 64) {
        // fill stage-B for k0+32 (overlaps stage-A WMMAs)
        a0B = load_frag(pa0, k0 + 32, half);
        a1B = load_frag(pa1, k0 + 32, half);
        bB0 = load_frag(pb[0], k0 + 32, half);
        bB1 = load_frag(pb[1], k0 + 32, half);
        bB2 = load_frag(pb[2], k0 + 32, half);
        bB3 = load_frag(pb[3], k0 + 32, half);
        WMMA_STEP(a0A, a1A, bA0, bA1, bA2, bA3);
        // refill stage-A for k0+64 (overlaps stage-B WMMAs)
        a0A = load_frag(pa0, k0 + 64, half);
        a1A = load_frag(pa1, k0 + 64, half);
        bA0 = load_frag(pb[0], k0 + 64, half);
        bA1 = load_frag(pb[1], k0 + 64, half);
        bA2 = load_frag(pb[2], k0 + 64, half);
        bA3 = load_frag(pb[3], k0 + 64, half);
        WMMA_STEP(a0B, a1B, bB0, bB1, bB2, bB3);
    }
    // last 64 columns of K
    a0B = load_frag(pa0, k0 + 32, half);
    a1B = load_frag(pa1, k0 + 32, half);
    bB0 = load_frag(pb[0], k0 + 32, half);
    bB1 = load_frag(pb[1], k0 + 32, half);
    bB2 = load_frag(pb[2], k0 + 32, half);
    bB3 = load_frag(pb[3], k0 + 32, half);
    WMMA_STEP(a0A, a1A, bA0, bA1, bA2, bA3);
    WMMA_STEP(a0B, a1B, bB0, bB1, bB2, bB3);
#undef WMMA_STEP

    // Epilogue. C/D layout: lane holds col n = lane&15; vgpr r holds row
    // m = tile_base + (lane>>4)*8 + r.
#pragma unroll
    for (int mi = 0; mi < 2; ++mi) {
#pragma unroll
        for (int j = 0; j < 4; ++j) {
            int n = ng * 64 + j * 16 + l15;
#pragma unroll
            for (int r = 0; r < 8; ++r) {
                int m = mt * 32 + mi * 16 + half * 8 + r;
                float v = acc[mi][j][r];
                if (bias)    v += bias[n];
                if (rowbias) v += rowbias[(long)(m / rowsPerBatch) * N + n];
                long off = (long)m * N + n;
                if (mode == 0) {
                    Cf32[off] = v;
                    if (Cf16) Cf16[off] = (_Float16)v;
                } else if (mode == 1) {
                    v = v > 0.f ? v : 0.f;
                    Cf32[off] = v;
                    if (Cf16) Cf16[off] = (_Float16)v;
                } else if (mode == 2) {
                    v = v > 0.f ? v : 0.f;
                    v = tanhf(v * gamma[n] + beta[n]);
                    Cf16[off] = (_Float16)v;
                } else if (mode == 3) {
                    Cf32[off] = v;
                } else { // mode 4: ARN residual
                    float o = Cf32[off] + tanhf(v);
                    Cf32[off] = o;
                    Cf16[off] = (_Float16)o;
                }
            }
        }
    }
}

// ---------------------------------------------------------------------------
// Per-sample Global LayerNorm over (valid time, feature); writes f16.
// ---------------------------------------------------------------------------
__global__ void k_gln(const float* __restrict__ aux, const int* __restrict__ aux_len,
                      const float* __restrict__ g, const float* __restrict__ bta,
                      _Float16* __restrict__ auxn16) {
    int b = blockIdx.x;
    int len = aux_len[b];
    const float* p = aux + (long)b * TA * FDIM;
    long nvalid = (long)len * FDIM;
    float s = 0.f, ss = 0.f;
    for (long i = threadIdx.x; i < nvalid; i += blockDim.x) {
        float x = p[i]; s += x; ss += x * x;
    }
    __shared__ float sh1[256], sh2[256];
    sh1[threadIdx.x] = s; sh2[threadIdx.x] = ss;
    __syncthreads();
    for (int st = 128; st > 0; st >>= 1) {
        if ((int)threadIdx.x < st) {
            sh1[threadIdx.x] += sh1[threadIdx.x + st];
            sh2[threadIdx.x] += sh2[threadIdx.x + st];
        }
        __syncthreads();
    }
    float mean = sh1[0] / (float)nvalid;
    float var  = sh2[0] / (float)nvalid - mean * mean;
    float inv  = rsqrtf(var + 1e-5f);
    _Float16* q = auxn16 + (long)b * TA * FDIM;
    for (long i = threadIdx.x; i < (long)TA * FDIM; i += blockDim.x) {
        int  f = (int)(i % FDIM);
        long t = i / FDIM;
        float v = (t < len) ? (g[f] * (p[i] - mean) * inv + bta[f]) : 0.f;
        q[i] = (_Float16)v;
    }
}

// Per-(b,f) mean / unbiased std over valid frames of h.
__global__ void k_stats(const float* __restrict__ h, const int* __restrict__ aux_len,
                        float* __restrict__ m_t, float* __restrict__ s_t) {
    int b = blockIdx.x, f = threadIdx.x;
    int len = aux_len[b];
    const float* p = h + (long)b * TA * HDIM + f;
    float s = 0.f;
    for (int t = 0; t < len; ++t) s += p[(long)t * HDIM];
    float mean = s / (float)len;
    float ss = 0.f;
    for (int t = 0; t < len; ++t) { float d = p[(long)t * HDIM] - mean; ss += d * d; }
    m_t[b * HDIM + f] = mean;
    s_t[b * HDIM + f] = sqrtf(fmaxf(ss / (float)(len - 1), 1e-4f));
}

// attbias[b][n] = att_b1[n] + m_t[b]·W1[256:512, n] + s_t[b]·W1[512:768, n]
__global__ void k_attbias(const float* __restrict__ m_t, const float* __restrict__ s_t,
                          const float* __restrict__ w1, const float* __restrict__ b1,
                          float* __restrict__ out) {
    int b = blockIdx.x, n = threadIdx.x;
    const float* wm = w1 + (long)HDIM * HDIM;
    const float* ws = w1 + (long)2 * HDIM * HDIM;
    float acc = b1[n];
    for (int k = 0; k < HDIM; ++k)
        acc += m_t[b * HDIM + k] * wm[(long)k * HDIM + n]
             + s_t[b * HDIM + k] * ws[(long)k * HDIM + n];
    out[b * HDIM + n] = acc;
}

// Masked softmax over time + attentive stats pooling + bn5, fused.
__global__ void k_pool(const float* __restrict__ logits, const float* __restrict__ h,
                       const int* __restrict__ aux_len,
                       const float* __restrict__ g5, const float* __restrict__ b5,
                       float* __restrict__ emb_pre) {
    int b = blockIdx.x, f = threadIdx.x;
    int len = aux_len[b];
    const float* pl = logits + (long)b * TA * HDIM + f;
    const float* ph = h      + (long)b * TA * HDIM + f;
    float mx = -1e30f;
    for (int t = 0; t < len; ++t) mx = fmaxf(mx, pl[(long)t * HDIM]);
    float se = 0.f, s1 = 0.f, s2 = 0.f;
    for (int t = 0; t < len; ++t) {
        float e  = __expf(pl[(long)t * HDIM] - mx);
        float hv = ph[(long)t * HDIM];
        se += e; s1 += e * hv; s2 += e * hv * hv;
    }
    float mu = s1 / se;
    float sg = sqrtf(fmaxf(s2 / se - mu * mu, 1e-4f));
    emb_pre[b * 512 + f]       = mu * g5[f]       + b5[f];
    emb_pre[b * 512 + 256 + f] = sg * g5[256 + f] + b5[256 + f];
}

// emb = (emb_pre @ fc6_w + fc6_b) * g6 + b6
__global__ void k_fc6(const float* __restrict__ emb_pre, const float* __restrict__ w,
                      const float* __restrict__ bb, const float* __restrict__ g6,
                      const float* __restrict__ b6, float* __restrict__ emb) {
    int b = blockIdx.x, j = threadIdx.x;
    float acc = bb[j];
    const float* e = emb_pre + b * 512;
    for (int k = 0; k < 512; ++k) acc += e[k] * w[(long)k * FDIM + j];
    emb[b * FDIM + j] = acc * g6[j] + b6[j];
}

// classify_out = (emb / max(||emb||,1e-12)) @ cls_w + cls_b
__global__ void k_classify(const float* __restrict__ emb, const float* __restrict__ cw,
                           const float* __restrict__ cb, float* __restrict__ cls_out) {
    int b = blockIdx.x;
    __shared__ float sh[256];
    __shared__ float inv;
    float x = emb[b * FDIM + threadIdx.x];
    sh[threadIdx.x] = x * x;
    __syncthreads();
    for (int st = 128; st > 0; st >>= 1) {
        if ((int)threadIdx.x < st) sh[threadIdx.x] += sh[threadIdx.x + st];
        __syncthreads();
    }
    if (threadIdx.x == 0) inv = 1.f / fmaxf(sqrtf(sh[0]), 1e-12f);
    __syncthreads();
    int j = threadIdx.x;
    if (j < NCLS) {
        float acc = cb[j];
        for (int k = 0; k < FDIM; ++k)
            acc += emb[b * FDIM + k] * inv * cw[(long)k * NCLS + j];
        cls_out[b * NCLS + j] = acc;
    }
}

// arnv[i][b][n] = arn_b[i][n] + emb[b] @ arn_w2[i][:, n]
__global__ void k_arnv(const float* __restrict__ emb, const float* __restrict__ w2,
                       const float* __restrict__ ab, float* __restrict__ v) {
    int b = blockIdx.x, i = blockIdx.y, n = threadIdx.x;
    const float* w = w2 + (long)i * FDIM * FDIM;
    float acc = ab[i * FDIM + n];
    const float* e = emb + b * FDIM;
    for (int k = 0; k < FDIM; ++k) acc += e[k] * w[(long)k * FDIM + n];
    v[((long)i * NB + b) * FDIM + n] = acc;
}

// Overlap-add with fold-of-ones denominator (1 for s<160, else 2).
__global__ void k_ola(const float* __restrict__ y, float* __restrict__ sig) {
    long i = (long)blockIdx.x * blockDim.x + threadIdx.x;
    if (i >= (long)NB * LSIG) return;
    int s = (int)(i % LSIG);
    int b = (int)(i / LSIG);
    int t1 = s / SHIFT;
    const float* p = y + (long)b * TIN * FRAME;
    float acc = p[(long)t1 * FRAME + (s - t1 * SHIFT)];
    float cnt = 1.f;
    if (t1 >= 1) {
        int t0 = t1 - 1;
        acc += p[(long)t0 * FRAME + (s - t0 * SHIFT)];
        cnt = 2.f;
    }
    sig[i] = acc / cnt;
}

// ---------------------------------------------------------------------------
extern "C" void kernel_launch(void* const* d_in, const int* in_sizes, int n_in,
                              void* d_out, int out_size, void* d_ws, size_t ws_size,
                              hipStream_t stream) {
    const float* input   = (const float*)d_in[0];
    const float* anchor  = (const float*)d_in[1];
    const int*   aux_len = (const int*)d_in[2];
    // d_in[3] = input_len (unused by the reference forward)
    const float* in_w    = (const float*)d_in[4];
    const float* in_b    = (const float*)d_in[5];
    const float* out_w   = (const float*)d_in[6];
    const float* out_b   = (const float*)d_in[7];
    const float* gln_g   = (const float*)d_in[8];
    const float* gln_b   = (const float*)d_in[9];
    const float* ecapa_w = (const float*)d_in[10];
    const float* ecapa_b = (const float*)d_in[11];
    const float* att_w1  = (const float*)d_in[12];
    const float* att_b1  = (const float*)d_in[13];
    const float* attbn_g = (const float*)d_in[14];
    const float* attbn_b = (const float*)d_in[15];
    const float* att_w2  = (const float*)d_in[16];
    const float* att_b2  = (const float*)d_in[17];
    const float* bn5_g   = (const float*)d_in[18];
    const float* bn5_b   = (const float*)d_in[19];
    const float* fc6_w   = (const float*)d_in[20];
    const float* fc6_b   = (const float*)d_in[21];
    const float* bn6_g   = (const float*)d_in[22];
    const float* bn6_b   = (const float*)d_in[23];
    const float* cls_w   = (const float*)d_in[24];
    const float* cls_b   = (const float*)d_in[25];
    const float* arn_w1  = (const float*)d_in[26];
    const float* arn_w2  = (const float*)d_in[27];
    const float* arn_b   = (const float*)d_in[28];

    float* sig_out = (float*)d_out;
    float* cls_out = (float*)d_out + (long)NB * LSIG;

    // ---- workspace carve-out (256B aligned) ----
    char* w = (char*)d_ws;
    auto alloc = [&](size_t bytes) -> char* {
        char* p = w;
        w += (bytes + 255) & ~(size_t)255;
        return p;
    };
    const int MIN = NB * TIN;   // 32000
    const int MAX_ = NB * TA;   // 12800
    _Float16* frames16 = (_Float16*)alloc((size_t)MIN  * FRAME * 2);
    _Float16* aux16    = (_Float16*)alloc((size_t)MAX_ * FRAME * 2);
    float*    out_f32  = (float*)   alloc((size_t)MIN  * FDIM  * 4);
    _Float16* X16a     = (_Float16*)alloc((size_t)MIN  * FDIM  * 2);
    _Float16* X16b     = (_Float16*)alloc((size_t)MIN  * FDIM  * 2);
    float*    aux_f32  = (float*)   alloc((size_t)MAX_ * FDIM  * 4);
    _Float16* auxn16   = (_Float16*)alloc((size_t)MAX_ * FDIM  * 2);
    float*    h_f32    = (float*)   alloc((size_t)MAX_ * HDIM  * 4);
    _Float16* h16      = (_Float16*)alloc((size_t)MAX_ * HDIM  * 2);
    _Float16* a16      = (_Float16*)alloc((size_t)MAX_ * HDIM  * 2);
    float*    logits   = (float*)   alloc((size_t)MAX_ * HDIM  * 4);
    float*    yF       = (float*)   alloc((size_t)MIN  * FRAME * 4);
    _Float16* in_wt    = (_Float16*)alloc((size_t)FDIM * FRAME * 2);
    _Float16* ecapa_wt = (_Float16*)alloc((size_t)HDIM * FDIM  * 2);
    _Float16* w1at     = (_Float16*)alloc((size_t)HDIM * HDIM  * 2);
    _Float16* w2t      = (_Float16*)alloc((size_t)HDIM * HDIM  * 2);
    _Float16* arnw1t   = (_Float16*)alloc((size_t)4 * FDIM * FDIM * 2);
    _Float16* outwt    = (_Float16*)alloc((size_t)FRAME * FDIM * 2);
    float*    m_t      = (float*)   alloc((size_t)NB * HDIM * 4);
    float*    s_t      = (float*)   alloc((size_t)NB * HDIM * 4);
    float*    attbias  = (float*)   alloc((size_t)NB * HDIM * 4);
    float*    emb_pre  = (float*)   alloc((size_t)NB * 512 * 4);
    float*    emb      = (float*)   alloc((size_t)NB * FDIM * 4);
    float*    arnv     = (float*)   alloc((size_t)4 * NB * FDIM * 4);

    auto gemm = [&](const _Float16* A, const _Float16* Bt, int M, int N, int K,
                    const float* bias, const float* rowbias, int rpb,
                    const float* g, const float* bt2,
                    float* cf, _Float16* ch, int mode) {
        int waves = (M / 32) * (N / 64);
        int blocks = (waves + 7) / 8;
        k_gemm_wmma<<<blocks, 256, 0, stream>>>(A, Bt, M, N, K, bias,
                                                rowbias, rpb, g, bt2, cf, ch, mode);
    };

    // 1. Frame signals into f16
    {
        long tot = (long)MIN * FRAME;
        k_frames<<<(int)((tot + 255) / 256), 256, 0, stream>>>(input, frames16, TIN, LSIG);
        tot = (long)MAX_ * FRAME;
        k_frames<<<(int)((tot + 255) / 256), 256, 0, stream>>>(anchor, aux16, TA, LAUX);
    }
    // 2. Transpose + convert weights
    auto tr = [&](const float* W, _Float16* Wt, int K, int N) {
        k_transpose_f16<<<(K * N + 255) / 256, 256, 0, stream>>>(W, Wt, K, N);
    };
    tr(in_w, in_wt, FRAME, FDIM);
    tr(ecapa_w, ecapa_wt, FDIM, HDIM);
    tr(att_w1, w1at, HDIM, HDIM);            // rows 0..255 of att_w1 only
    tr(att_w2, w2t, HDIM, HDIM);
    for (int i = 0; i < 4; ++i)
        tr(arn_w1 + (long)i * FDIM * FDIM, arnw1t + (long)i * FDIM * FDIM, FDIM, FDIM);
    tr(out_w, outwt, FDIM, FRAME);

    // 3. Encoder GEMMs
    gemm(frames16, in_wt, MIN,  FDIM, FRAME, in_b, nullptr, 1, nullptr, nullptr,
         out_f32, X16a, 0);                                     // out + f16 copy
    gemm(aux16,    in_wt, MAX_, FDIM, FRAME, in_b, nullptr, 1, nullptr, nullptr,
         aux_f32, nullptr, 3);                                  // aux (f32)

    // 4. Anchor branch
    k_gln<<<NB, 256, 0, stream>>>(aux_f32, aux_len, gln_g, gln_b, auxn16);
    gemm(auxn16, ecapa_wt, MAX_, HDIM, FDIM, ecapa_b, nullptr, 1, nullptr, nullptr,
         h_f32, h16, 1);                                        // h = relu(...)
    k_stats<<<NB, HDIM, 0, stream>>>(h_f32, aux_len, m_t, s_t);
    k_attbias<<<NB, HDIM, 0, stream>>>(m_t, s_t, att_w1, att_b1, attbias);
    gemm(h16, w1at, MAX_, HDIM, HDIM, nullptr, attbias, TA, attbn_g, attbn_b,
         nullptr, a16, 2);                                      // a = tanh(bn(relu))
    gemm(a16, w2t, MAX_, HDIM, HDIM, att_b2, nullptr, 1, nullptr, nullptr,
         logits, nullptr, 3);                                   // logits
    k_pool<<<NB, HDIM, 0, stream>>>(logits, h_f32, aux_len, bn5_g, bn5_b, emb_pre);
    k_fc6<<<NB, FDIM, 0, stream>>>(emb_pre, fc6_w, fc6_b, bn6_g, bn6_b, emb);
    k_classify<<<NB, 256, 0, stream>>>(emb, cls_w, cls_b, cls_out);

    // 5. ARN residual blocks
    k_arnv<<<dim3(NB, 4), FDIM, 0, stream>>>(emb, arn_w2, arn_b, arnv);
    _Float16* xin  = X16a;
    _Float16* xout = X16b;
    for (int i = 0; i < 4; ++i) {
        gemm(xin, arnw1t + (long)i * FDIM * FDIM, MIN, FDIM, FDIM,
             nullptr, arnv + (long)i * NB * FDIM, TIN, nullptr, nullptr,
             out_f32, xout, 4);
        _Float16* t = xin; xin = xout; xout = t;
    }

    // 6. Decode + overlap-add
    gemm(xin, outwt, MIN, FRAME, FDIM, out_b, nullptr, 1, nullptr, nullptr,
         yF, nullptr, 3);
    {
        long tot = (long)NB * LSIG;
        k_ola<<<(int)((tot + 255) / 256), 256, 0, stream>>>(yF, sig_out);
    }
}
